// GINBackboneVN_33921651703944
// MI455X (gfx1250) — compile-verified, matrix operand order checked by reading, not run on hardware
//
#include <hip/hip_runtime.h>

typedef __attribute__((ext_vector_type(16))) __bf16 v16bf;
typedef __attribute__((ext_vector_type(8)))  float  v8f;

#define DD    300
#define NF4   75      /* DD/4 */
#define GNUM  4000
#define LNUM  5
#define EPSV  1e-5f
#define KP1   320     /* padded K for D=300 */
#define NP1   640     /* padded Nout for 600 */
#define KP2   608     /* padded K for 600 */
#define NP2   320     /* padded Nout for 300 */

// native bf16 converts (hardware v_cvt on gfx1250, RNE)
static __device__ __forceinline__ unsigned short f2bf(float x) {
  union { __bf16 b; unsigned short u; } c;
  c.b = (__bf16)x;
  return c.u;
}
static __device__ __forceinline__ unsigned int pk2bf(float lo, float hi) {
  union { __bf16 b[2]; unsigned int u; } c;
  c.b[0] = (__bf16)lo;
  c.b[1] = (__bf16)hi;
  return c.u;
}

union Frag {
  unsigned int u[8];
  v16bf v;
};

// ---------------------------------------------------------------------------
// Tiled WMMA GEMM:  Out[M,Nout] = epilogue( A[M,Kp] @ Wt^T )
//   Wt: bf16, pre-transposed+padded [Np][Kp] (Np mult of 64, Kp mult of 32)
//   A:  lda = Kp (pad columns zeroed by caller)  -> NO bounds checks in K loop
//   A rows handled by clamping (dead rows never stored) -> no exec masking
// 128x64 block tile, 4 waves (wave32), each wave 32x64 via 2x4 WMMA 16x16x32.
// Staging issues all global loads before converting/storing to LDS.
// ---------------------------------------------------------------------------
template <bool ABF16, bool OBF16, bool RELU, bool BN>
__global__ __launch_bounds__(128) void k_gemm(
    const void* __restrict__ Ap, int lda, int M, int Kp,
    const unsigned short* __restrict__ Wt, int ldb, int Nout,
    const float* __restrict__ bias,
    const float* __restrict__ bng, const float* __restrict__ bnb,
    const float* __restrict__ bnm, const float* __restrict__ bnv,
    void* __restrict__ Op, int ldo)
{
  __shared__ unsigned short sA[128 * 40];  // [m][k], stride 40 (conflict-free)
  __shared__ unsigned short sB[64 * 40];   // [n][k]
  const int t    = threadIdx.x;
  const int m0   = blockIdx.x * 128;
  const int n0   = blockIdx.y * 64;
  const int lane = t & 31;
  const int wid  = t >> 5;
  const int wm   = wid * 32;
  const int hf   = lane >> 4;              // half-wave selects K-group
  const int l15  = lane & 15;

  v8f acc[2][4] = {};

  for (int kk = 0; kk < Kp; kk += 32) {
    uint4 wb[2];
    // ---------- phase 1: issue ALL global loads ----------
    #pragma unroll
    for (int p = 0; p < 2; ++p) {
      const int lin = t + p * 128;                      // 64 rows x 4 uint4
      const int r = lin >> 2, c = lin & 3;
      wb[p] = *(const uint4*)(Wt + (size_t)(n0 + r) * ldb + kk + c * 8);
    }
    if (ABF16) {
      const unsigned short* A = (const unsigned short*)Ap;
      uint4 va[4];
      #pragma unroll
      for (int p = 0; p < 4; ++p) {                     // 128 rows x 4 uint4
        const int lin = t + p * 128;
        const int r = lin >> 2, c = lin & 3;
        const int row = min(m0 + r, M - 1);
        va[p] = *(const uint4*)(A + (size_t)row * lda + kk + c * 8);
      }
      if (kk + 32 < Kp)
        __builtin_prefetch((const char*)Ap +
            ((size_t)min(m0 + (t & 127), M - 1) * lda + kk + 32) * 2, 0, 0);
      // ---------- phase 2: store to LDS ----------
      #pragma unroll
      for (int p = 0; p < 4; ++p) {
        const int lin = t + p * 128;
        const int r = lin >> 2, c = lin & 3;
        *(uint4*)(&sA[r * 40 + c * 8]) = va[p];         // byte r*80+c*16: 16B ok
      }
    } else {
      const float* A = (const float*)Ap;
      float4 fa8[8];
      #pragma unroll
      for (int p = 0; p < 8; ++p) {                     // 128 rows x 8 float4
        const int lin = t + p * 128;
        const int r = lin >> 3, c = lin & 7;
        const int row = min(m0 + r, M - 1);
        fa8[p] = *(const float4*)(A + (size_t)row * lda + kk + c * 4);
      }
      if (kk + 32 < Kp)
        __builtin_prefetch((const char*)Ap +
            ((size_t)min(m0 + (t & 127), M - 1) * lda + kk + 32) * 4, 0, 0);
      // ---------- phase 2: convert + store to LDS ----------
      #pragma unroll
      for (int p = 0; p < 8; ++p) {
        const int lin = t + p * 128;
        const int r = lin >> 3, c = lin & 7;
        uint2 pk;
        pk.x = pk2bf(fa8[p].x, fa8[p].y);
        pk.y = pk2bf(fa8[p].z, fa8[p].w);
        *(uint2*)(&sA[r * 40 + c * 4]) = pk;            // byte r*80+c*8: 8B ok
      }
    }
    #pragma unroll
    for (int p = 0; p < 2; ++p) {
      const int lin = t + p * 128;
      const int r = lin >> 2, c = lin & 3;
      *(uint4*)(&sB[r * 40 + c * 8]) = wb[p];
    }
    __syncthreads();

    // ---- fragments: 16-bit A layout (lanes 0-15: K0-7/16-23, 16-31: K8-15/24-31)
    Frag fa[2], fb[4];
    #pragma unroll
    for (int ms = 0; ms < 2; ++ms) {
      const unsigned short* base = &sA[(wm + ms * 16 + l15) * 40 + hf * 8];
      #pragma unroll
      for (int i = 0; i < 4; ++i) {
        fa[ms].u[i]     = *(const unsigned int*)(base + 2 * i);
        fa[ms].u[4 + i] = *(const unsigned int*)(base + 16 + 2 * i);
      }
    }
    #pragma unroll
    for (int ns = 0; ns < 4; ++ns) {
      const unsigned short* base = &sB[(ns * 16 + l15) * 40 + hf * 8];
      #pragma unroll
      for (int i = 0; i < 4; ++i) {
        fb[ns].u[i]     = *(const unsigned int*)(base + 2 * i);
        fb[ns].u[4 + i] = *(const unsigned int*)(base + 16 + 2 * i);
      }
    }
    #pragma unroll
    for (int ms = 0; ms < 2; ++ms)
      #pragma unroll
      for (int ns = 0; ns < 4; ++ns)
        acc[ms][ns] = __builtin_amdgcn_wmma_f32_16x16x32_bf16(
            false, fa[ms].v, false, fb[ns].v, (short)0, acc[ms][ns], false, false);
    __syncthreads();
  }

  // ---- epilogue: C f32 16x16 -> lane = col + 16*(row>=8), vgpr i = row%8
  #pragma unroll
  for (int ns = 0; ns < 4; ++ns) {
    const int col = n0 + ns * 16 + l15;
    if (col >= Nout) continue;
    const float bv = bias ? bias[col] : 0.f;
    float sc = 1.f, sh = 0.f;
    if (BN) {
      sc = bng[col] * rsqrtf(bnv[col] + EPSV);
      sh = bnb[col] - bnm[col] * sc;
    }
    #pragma unroll
    for (int ms = 0; ms < 2; ++ms) {
      const int rb = m0 + wm + ms * 16 + hf * 8;
      #pragma unroll
      for (int i = 0; i < 8; ++i) {
        const int row = rb + i;
        if (row >= M) continue;
        float v = acc[ms][ns][i] + bv;
        if (BN) v = v * sc + sh;
        if (RELU) v = fmaxf(v, 0.f);
        if (OBF16) ((unsigned short*)Op)[(size_t)row * ldo + col] = f2bf(v);
        else       ((float*)Op)[(size_t)row * ldo + col] = v;
      }
    }
  }
}

// ---------------------------------------------------------------------------
// Weight prep: Wt[n][k] = bf16(W[k][n]), zero-padded to [Np][Kp]. One-time.
// ---------------------------------------------------------------------------
__global__ void k_w_prep(unsigned short* __restrict__ Wt, const float* __restrict__ W,
                         int K, int Nout, int Kp, int Np) {
  const long idx = (long)blockIdx.x * blockDim.x + threadIdx.x;
  if (idx >= (long)Np * Kp) return;
  const int n = (int)(idx / Kp);
  const int k = (int)(idx - (long)n * Kp);
  const float v = (k < K && n < Nout) ? W[(size_t)k * Nout + n] : 0.f;
  Wt[idx] = f2bf(v);
}

// zero pad columns [c0,c0+nc) of a matrix with row stride ld
__global__ void k_zero_pad_f32(float* __restrict__ p, int rows, int ld, int c0, int nc) {
  const long idx = (long)blockIdx.x * blockDim.x + threadIdx.x;
  if (idx >= (long)rows * nc) return;
  const int r = (int)(idx / nc), c = (int)(idx - (long)r * nc);
  p[(size_t)r * ld + c0 + c] = 0.f;
}
__global__ void k_zero_pad_bf16(unsigned short* __restrict__ p, int rows, int ld, int c0, int nc) {
  const long idx = (long)blockIdx.x * blockDim.x + threadIdx.x;
  if (idx >= (long)rows * nc) return;
  const int r = (int)(idx / nc), c = (int)(idx - (long)r * nc);
  p[(size_t)r * ld + c0 + c] = 0;
}

// ---------------------------------------------------------------------------
// GIN message pass: aggr[dst[e]] += h[src[e]] + e1[etype[e]] + e2[edir[e]]
// (aggr has row stride KP1; only cols < DD touched)
// ---------------------------------------------------------------------------
__global__ __launch_bounds__(256) void k_edge_scatter(
    const float* __restrict__ h, const int* __restrict__ src,
    const int* __restrict__ dst, const int* __restrict__ et,
    const int* __restrict__ ed, const float* __restrict__ e1,
    const float* __restrict__ e2, float* __restrict__ aggr, int E)
{
  __shared__ float s1[6 * DD];
  __shared__ float s2[3 * DD];
  __shared__ int ss[32], sd[32], so1[32], so2[32];
  const int t = threadIdx.x;
  for (int i = t; i < 6 * DD; i += 256) s1[i] = e1[i];
  for (int i = t; i < 3 * DD; i += 256) s2[i] = e2[i];
  const int e0 = blockIdx.x * 32;
  if (t < 32) {
    const int e = e0 + t;
    if (e < E) {
      ss[t] = src[e]; sd[t] = dst[e];
      so1[t] = et[e] * DD; so2[t] = ed[e] * DD;
    } else {
      ss[t] = -1;
    }
  }
  __syncthreads();
  for (int j = t; j < 32 * NF4; j += 256) {
    const int e = j / NF4;
    const int q = (j - e * NF4) * 4;
    const int s = ss[e];
    if (s < 0) continue;
    const float4 hv = *(const float4*)(h + (size_t)s * DD + q);
    float* ap = aggr + (size_t)sd[e] * KP1 + q;
    const float* p1 = s1 + so1[e] + q;
    const float* p2 = s2 + so2[e] + q;
    atomicAdd(ap + 0, hv.x + p1[0] + p2[0]);
    atomicAdd(ap + 1, hv.y + p1[1] + p2[1]);
    atomicAdd(ap + 2, hv.z + p1[2] + p2[2]);
    atomicAdd(ap + 3, hv.w + p1[3] + p2[3]);
  }
}

// --------------------------- elementwise helpers ---------------------------
__global__ void k_zero(float* __restrict__ p, long n) {
  const long i = (long)blockIdx.x * blockDim.x + threadIdx.x;
  if (i < n) p[i] = 0.f;
}

__global__ void k_count(float* __restrict__ counts, const int* __restrict__ batch, int N) {
  const int i = blockIdx.x * blockDim.x + threadIdx.x;
  if (i < N) atomicAdd(&counts[batch[i]], 1.f);
}

__global__ void k_vnh_init(float* __restrict__ vnh, const float* __restrict__ vne, long n) {
  const long i = (long)blockIdx.x * blockDim.x + threadIdx.x;
  if (i < n) {
    const int g = (int)(i / DD);
    const int d = (int)(i - (long)g * DD);
    vnh[(size_t)g * KP1 + d] = vne[d];
  }
}

// h = emb1[x_atom] + emb2[x_chir] + vn_emb  (initial vn_h[batch] == vn_emb)
__global__ void k_hinit(float* __restrict__ h, const int* __restrict__ xa,
                        const int* __restrict__ xc, const float* __restrict__ e1,
                        const float* __restrict__ e2, const float* __restrict__ vne, int N) {
  const long idx = (long)blockIdx.x * blockDim.x + threadIdx.x;
  if (idx >= (long)N * NF4) return;
  const int i = (int)(idx / NF4);
  const int q = (int)(idx - (long)i * NF4) * 4;
  const float4 a = *(const float4*)(e1 + (size_t)xa[i] * DD + q);
  const float4 b = *(const float4*)(e2 + (size_t)xc[i] * DD + q);
  const float4 v = *(const float4*)(vne + q);
  float4 o;
  o.x = a.x + b.x + v.x; o.y = a.y + b.y + v.y;
  o.z = a.z + b.z + v.z; o.w = a.w + b.w + v.w;
  *(float4*)(h + (size_t)i * DD + q) = o;
}

// aggr = h + (e1[row4] + e2[row0])  (self-loop; accumulator init; stride KP1)
__global__ void k_aggr_init(float* __restrict__ aggr, const float* __restrict__ h,
                            const float* __restrict__ e1, const float* __restrict__ e2, int N) {
  const long idx = (long)blockIdx.x * blockDim.x + threadIdx.x;
  if (idx >= (long)N * NF4) return;
  const int i = (int)(idx / NF4);
  const int q = (int)(idx - (long)i * NF4) * 4;
  const float4 hv = *(const float4*)(h + (size_t)i * DD + q);
  const float4 sa = *(const float4*)(e1 + 4 * DD + q);
  const float4 sb = *(const float4*)(e2 + q);
  float4 o;
  o.x = hv.x + sa.x + sb.x; o.y = hv.y + sa.y + sb.y;
  o.z = hv.z + sa.z + sb.z; o.w = hv.w + sa.w + sb.w;
  *(float4*)(aggr + (size_t)i * KP1 + q) = o;
}

__global__ void k_pool_scatter(float* __restrict__ pooled, const float* __restrict__ z,
                               const int* __restrict__ batch, int N) {
  const long idx = (long)blockIdx.x * blockDim.x + threadIdx.x;
  if (idx >= (long)N * NF4) return;
  const int i = (int)(idx / NF4);
  const int q = (int)(idx - (long)i * NF4) * 4;
  const float4 zv = *(const float4*)(z + (size_t)i * DD + q);
  float* p = pooled + (size_t)batch[i] * KP1 + q;
  atomicAdd(p + 0, zv.x);
  atomicAdd(p + 1, zv.y);
  atomicAdd(p + 2, zv.z);
  atomicAdd(p + 3, zv.w);
}

__global__ void k_vn_new(float* __restrict__ out, const float* __restrict__ pooled,
                         const float* __restrict__ counts, const float* __restrict__ vnh, long n) {
  const long i = (long)blockIdx.x * blockDim.x + threadIdx.x;
  if (i < n) {
    const int g = (int)(i / DD);
    const int d = (int)(i - (long)g * DD);
    const size_t o = (size_t)g * KP1 + d;
    out[o] = pooled[o] / fmaxf(counts[g], 1.f) + vnh[o];
  }
}

// row-wise LayerNorm (population var) + ReLU, in place (stride KP1, cols < DD)
__global__ __launch_bounds__(128) void k_ln_relu(float* __restrict__ tmat,
                                                 const float* __restrict__ g,
                                                 const float* __restrict__ b) {
  __shared__ float red[128];
  const int t = threadIdx.x;
  float* row = tmat + (size_t)blockIdx.x * KP1;
  float s = 0.f;
  for (int i = t; i < DD; i += 128) s += row[i];
  red[t] = s; __syncthreads();
  for (int o = 64; o > 0; o >>= 1) { if (t < o) red[t] += red[t + o]; __syncthreads(); }
  const float mu = red[0] / DD;
  __syncthreads();
  float v = 0.f;
  for (int i = t; i < DD; i += 128) { const float d = row[i] - mu; v += d * d; }
  red[t] = v; __syncthreads();
  for (int o = 64; o > 0; o >>= 1) { if (t < o) red[t] += red[t + o]; __syncthreads(); }
  const float rs = rsqrtf(red[0] / DD + EPSV);
  for (int i = t; i < DD; i += 128) {
    const float x = (row[i] - mu) * rs * g[i] + b[i];
    row[i] = fmaxf(x, 0.f);
  }
}

__global__ void k_add_vn(float* __restrict__ z, const float* __restrict__ vnh,
                         const int* __restrict__ batch, int N) {
  const long idx = (long)blockIdx.x * blockDim.x + threadIdx.x;
  if (idx >= (long)N * NF4) return;
  const int i = (int)(idx / NF4);
  const int q = (int)(idx - (long)i * NF4) * 4;
  float4 a = *(const float4*)(z + (size_t)i * DD + q);
  const float4 b = *(const float4*)(vnh + (size_t)batch[i] * KP1 + q);
  a.x += b.x; a.y += b.y; a.z += b.z; a.w += b.w;
  *(float4*)(z + (size_t)i * DD + q) = a;
}

// ---------------------------------------------------------------------------
extern "C" void kernel_launch(void* const* d_in, const int* in_sizes, int n_in,
                              void* d_out, int out_size, void* d_ws, size_t ws_size,
                              hipStream_t stream) {
  (void)n_in; (void)out_size; (void)ws_size;
  const int*   x_atom  = (const int*)d_in[0];
  const int*   x_chir  = (const int*)d_in[1];
  const int*   src     = (const int*)d_in[2];
  const int*   dstp    = (const int*)d_in[3];
  const int*   etype   = (const int*)d_in[4];
  const int*   edir    = (const int*)d_in[5];
  const int*   batch   = (const int*)d_in[6];
  const float* x_emb1  = (const float*)d_in[7];
  const float* x_emb2  = (const float*)d_in[8];
  const float* edge_e1 = (const float*)d_in[9];
  const float* edge_e2 = (const float*)d_in[10];
  const float* mlp_w1  = (const float*)d_in[11];
  const float* mlp_b1  = (const float*)d_in[12];
  const float* mlp_w2  = (const float*)d_in[13];
  const float* mlp_b2  = (const float*)d_in[14];
  const float* bn_g    = (const float*)d_in[15];
  const float* bn_b    = (const float*)d_in[16];
  const float* bn_m    = (const float*)d_in[17];
  const float* bn_v    = (const float*)d_in[18];
  const float* vn_emb  = (const float*)d_in[19];
  const float* vn_w1   = (const float*)d_in[20];
  const float* vn_b1   = (const float*)d_in[21];
  const float* vn_ln_g = (const float*)d_in[22];
  const float* vn_ln_b = (const float*)d_in[23];
  const float* vn_w2   = (const float*)d_in[24];
  const float* vn_b2   = (const float*)d_in[25];

  const int N = in_sizes[0];
  const int E = in_sizes[2];

  char* ws = (char*)d_ws;
  auto algn = [](size_t x) { return (x + 255) & ~(size_t)255; };
  size_t off = 0;
  float*          aggr   = (float*)(ws + off);          off += algn((size_t)N * KP1 * 4);
  unsigned short* z1     = (unsigned short*)(ws + off); off += algn((size_t)N * KP2 * 2);
  float*          vnh    = (float*)(ws + off);          off += algn((size_t)GNUM * KP1 * 4);
  float*          pooled = (float*)(ws + off);          off += algn((size_t)GNUM * KP1 * 4);
  float*          vnn    = (float*)(ws + off);          off += algn((size_t)GNUM * KP1 * 4);
  float*          vnt    = (float*)(ws + off);          off += algn((size_t)GNUM * KP1 * 4);
  float*          counts = (float*)(ws + off);          off += algn((size_t)GNUM * 4);
  unsigned short* w1t    = (unsigned short*)(ws + off); off += algn((size_t)LNUM * NP1 * KP1 * 2);
  unsigned short* w2t    = (unsigned short*)(ws + off); off += algn((size_t)LNUM * NP2 * KP2 * 2);
  unsigned short* vw1t   = (unsigned short*)(ws + off); off += algn((size_t)(LNUM-1) * NP2 * KP1 * 2);
  unsigned short* vw2t   = (unsigned short*)(ws + off); off += algn((size_t)(LNUM-1) * NP2 * KP1 * 2);

  float* h = (float*)d_out;   // node features updated in place; final layer = output

  auto cdiv = [](long a, long b) { return (int)((a + b - 1) / b); };
  const long NW = (long)N * NF4;
  const long GW = (long)GNUM * DD;

  // ---- one-time prep: counts, vn_h, weight transpose/pad, pad-zeroing ----
  k_zero<<<cdiv(GNUM, 256), 256, 0, stream>>>(counts, (long)GNUM);
  k_count<<<cdiv(N, 256), 256, 0, stream>>>(counts, batch, N);
  k_zero<<<cdiv((long)GNUM * KP1, 256), 256, 0, stream>>>(vnh, (long)GNUM * KP1);
  k_zero<<<cdiv((long)GNUM * KP1, 256), 256, 0, stream>>>(vnn, (long)GNUM * KP1);
  k_zero<<<cdiv((long)GNUM * KP1, 256), 256, 0, stream>>>(vnt, (long)GNUM * KP1);
  k_vnh_init<<<cdiv(GW, 256), 256, 0, stream>>>(vnh, vn_emb, GW);
  k_zero_pad_f32<<<cdiv((long)N * (KP1 - DD), 256), 256, 0, stream>>>(aggr, N, KP1, DD, KP1 - DD);
  k_zero_pad_bf16<<<cdiv((long)N * (KP2 - 2 * DD), 256), 256, 0, stream>>>(z1, N, KP2, 2 * DD, KP2 - 2 * DD);
  for (int l = 0; l < LNUM; ++l) {
    k_w_prep<<<cdiv((long)NP1 * KP1, 256), 256, 0, stream>>>(
        w1t + (size_t)l * NP1 * KP1, mlp_w1 + (size_t)l * DD * 2 * DD, DD, 2 * DD, KP1, NP1);
    k_w_prep<<<cdiv((long)NP2 * KP2, 256), 256, 0, stream>>>(
        w2t + (size_t)l * NP2 * KP2, mlp_w2 + (size_t)l * 2 * DD * DD, 2 * DD, DD, KP2, NP2);
    if (l < LNUM - 1) {
      k_w_prep<<<cdiv((long)NP2 * KP1, 256), 256, 0, stream>>>(
          vw1t + (size_t)l * NP2 * KP1, vn_w1 + (size_t)l * DD * DD, DD, DD, KP1, NP2);
      k_w_prep<<<cdiv((long)NP2 * KP1, 256), 256, 0, stream>>>(
          vw2t + (size_t)l * NP2 * KP1, vn_w2 + (size_t)l * DD * DD, DD, DD, KP1, NP2);
    }
  }
  k_hinit<<<cdiv(NW, 256), 256, 0, stream>>>(h, x_atom, x_chir, x_emb1, x_emb2, vn_emb, N);

  // ---- layers ----
  for (int l = 0; l < LNUM; ++l) {
    const float* e1l = edge_e1 + (size_t)l * 6 * DD;
    const float* e2l = edge_e2 + (size_t)l * 3 * DD;

    k_aggr_init<<<cdiv(NW, 256), 256, 0, stream>>>(aggr, h, e1l, e2l, N);
    k_edge_scatter<<<cdiv(E, 32), 256, 0, stream>>>(h, src, dstp, etype, edir,
                                                    e1l, e2l, aggr, E);

    // GEMM1: z1 = relu(aggr @ W1 + b1)  -> bf16 (stride KP2)
    dim3 g1(cdiv(N, 128), NP1 / 64);
    k_gemm<false, true, true, false><<<g1, 128, 0, stream>>>(
        aggr, KP1, N, KP1, w1t + (size_t)l * NP1 * KP1, KP1, 2 * DD,
        mlp_b1 + (size_t)l * 2 * DD, nullptr, nullptr, nullptr, nullptr, z1, KP2);

    // GEMM2: h = BN(z1 @ W2 + b2) [+ relu if not last]
    dim3 g2(cdiv(N, 128), NP2 / 64);
    if (l < LNUM - 1) {
      k_gemm<true, false, true, true><<<g2, 128, 0, stream>>>(
          z1, KP2, N, KP2, w2t + (size_t)l * NP2 * KP2, KP2, DD,
          mlp_b2 + (size_t)l * DD, bn_g + (size_t)l * DD, bn_b + (size_t)l * DD,
          bn_m + (size_t)l * DD, bn_v + (size_t)l * DD, h, DD);

      // virtual-node update
      k_zero<<<cdiv((long)GNUM * KP1, 256), 256, 0, stream>>>(pooled, (long)GNUM * KP1);
      k_pool_scatter<<<cdiv(NW, 256), 256, 0, stream>>>(pooled, h, batch, N);
      k_vn_new<<<cdiv(GW, 256), 256, 0, stream>>>(vnn, pooled, counts, vnh, GW);

      dim3 gv(cdiv(GNUM, 128), NP2 / 64);
      k_gemm<false, false, false, false><<<gv, 128, 0, stream>>>(
          vnn, KP1, GNUM, KP1, vw1t + (size_t)l * NP2 * KP1, KP1, DD,
          vn_b1 + (size_t)l * DD, nullptr, nullptr, nullptr, nullptr, vnt, KP1);
      k_ln_relu<<<GNUM, 128, 0, stream>>>(vnt, vn_ln_g + (size_t)l * DD,
                                          vn_ln_b + (size_t)l * DD);
      k_gemm<false, false, false, false><<<gv, 128, 0, stream>>>(
          vnt, KP1, GNUM, KP1, vw2t + (size_t)l * NP2 * KP1, KP1, DD,
          vn_b2 + (size_t)l * DD, nullptr, nullptr, nullptr, nullptr, vnh, KP1);
      k_add_vn<<<cdiv(NW, 256), 256, 0, stream>>>(h, vnh, batch, N);
    } else {
      k_gemm<true, false, false, true><<<g2, 128, 0, stream>>>(
          z1, KP2, N, KP2, w2t + (size_t)l * NP2 * KP2, KP2, DD,
          mlp_b2 + (size_t)l * DD, bn_g + (size_t)l * DD, bn_b + (size_t)l * DD,
          bn_m + (size_t)l * DD, bn_v + (size_t)l * DD, h, DD);
    }
  }
}
